// NTLBGCore_32882269618908
// MI455X (gfx1250) — matrix operand order repaired and checked by reading, not compiled, not run on hardware
//
#include <hip/hip_runtime.h>

typedef __attribute__((ext_vector_type(16))) _Float16 v16h;
typedef __attribute__((ext_vector_type(8)))  _Float16 v8h;
typedef __attribute__((ext_vector_type(8)))  float    v8f;

#define B_     8
#define T_     8192
#define DV     1024
#define DQ     768
#define KREP   6
#define NHEADS 8
#define HD     128
#define CAND   18

// d_out flat layout (floats): refined, idx, dist, mu_q, sigma_q, attn_w
#define REFINED_OFF 0
#define IDX_OFF     49152
#define DIST_OFF    49200
#define MU_OFF      114736
#define SG_OFF      122928
#define ATTN_OFF    131120

// ---------------------------------------------------------------- utilities

__device__ __forceinline__ float wave_sum(float v) {
#pragma unroll
  for (int m = 16; m > 0; m >>= 1) v += __shfl_xor(v, m, 32);
  return v;
}

// A fragment (16x32 f16) from a row-major f16 LDS source that is ZERO-PADDED
// to 16 rows. ISA layout per lane (hh = lane/16, r = lane%16):
//   VGPR v<4  -> K = hh*8 + 2v,+1    (contiguous run  hh*8   .. hh*8+7)
//   VGPR v>=4 -> K = 16+hh*8+2(v-4) (contiguous run 16+hh*8 .. 16+hh*8+7)
// => exactly two 16-byte vector loads, branch-free.
__device__ __forceinline__ v16h a_frag_lds(const _Float16* A, int ld, int kbase) {
  int lane = threadIdx.x & 31;
  int hh = lane >> 4, r = lane & 15;
  const _Float16* p = A + r * ld + kbase + hh * 8;
  v8h lo = *(const v8h*)(p);
  v8h hi = *(const v8h*)(p + 16);
  v16h a;
#pragma unroll
  for (int i = 0; i < 8; ++i) { a[i] = lo[i]; a[8 + i] = hi[i]; }
  return a;
}

// B fragment (32x16 f16) for y = x @ W.T : B(k,n) = W[n*ldk + k] (f32 -> f16).
// ISA layout: lanes 0-15 -> K=0..7,16..23; lanes 16-31 -> K=8..15,24..31.
// two runs of 8 consecutive f32 per lane (+0..7, +16..23) -> 4x global_load_b128 + packed converts.
__device__ __forceinline__ v16h b_frag_w(const float* W, int ldk, int n, int kbase) {
  int lane = threadIdx.x & 31;
  int hh = lane >> 4;
  const float* p = W + n * ldk + kbase + hh * 8;
  v16h b;
#pragma unroll
  for (int v = 0; v < 16; ++v) b[v] = (_Float16)p[v < 8 ? v : v + 8];
  return b;
}

// ------------------------------------------------- kernel 1: MLP + LN (WMMA)
// One block (8 waves). out = Linear2(ReLU(LN(Linear1(x)))) for 8 rows.
// sigma_mode: out = softplus(.)+1e-6, ws = 1/out ; else out = ws = value.
__global__ void __launch_bounds__(256)
mlp_ln_wmma(const float* __restrict__ x,   // [8, DQ]
            const float* __restrict__ w1,  // [DV, DQ]
            const float* __restrict__ b1,
            const float* __restrict__ g,
            const float* __restrict__ bt,
            const float* __restrict__ w2,  // [DV, DV]
            const float* __restrict__ b2,
            float* __restrict__ out,       // [8, DV]
            float* __restrict__ wsv,       // [8, DV]
            int sigma_mode) {
  __shared__ __align__(16) _Float16 x16[16][DQ];  // 24 KB (rows 8-15 zero)
  __shared__ __align__(16) _Float16 H16[16][DV];  // 32 KB (rows 8-15 zero)
  int tid = threadIdx.x;
  int w = tid >> 5, lane = tid & 31;

  // stage x into LDS f16, zero-pad rows 8..15 of both A buffers
  for (int i = tid; i < 8 * DQ; i += 256) (&x16[8][0])[i] = (_Float16)0.f;
  for (int i = tid; i < 8 * DV; i += 256) (&H16[8][0])[i] = (_Float16)0.f;
  for (int i = tid; i < 8 * DQ; i += 256) (&x16[0][0])[i] = (_Float16)x[i];
  __syncthreads();

  // Stage A: H = x @ w1.T   (M=16 padded, N=1024, K=768)
  for (int t = w; t < DV / 16; t += 8) {
    v8f acc = {};
    for (int kc = 0; kc < DQ / 32; ++kc) {
      v16h af = a_frag_lds(&x16[0][0], DQ, kc * 32);
      v16h bf = b_frag_w(w1, DQ, t * 16 + (lane & 15), kc * 32);
      acc = __builtin_amdgcn_wmma_f32_16x16x32_f16(false, af, false, bf, (short)0, acc, false, false);
    }
    if (lane < 16) {
      int n = t * 16 + lane;
      float bias = b1[n];
#pragma unroll
      for (int j = 0; j < 8; ++j) H16[j][n] = (_Float16)(acc[j] + bias);  // lanes 0-15 hold M=j
    }
  }
  __syncthreads();

  // LayerNorm + ReLU in place; wave w handles row w
  {
    float s1 = 0.f, s2 = 0.f;
    for (int c = lane; c < DV; c += 32) { float h = (float)H16[w][c]; s1 += h; s2 += h * h; }
    s1 = wave_sum(s1); s2 = wave_sum(s2);
    float mean = s1 * (1.f / DV);
    float var  = s2 * (1.f / DV) - mean * mean;
    float rstd = rsqrtf(var + 1e-5f);
    for (int c = lane; c < DV; c += 32) {
      float h = ((float)H16[w][c] - mean) * rstd * g[c] + bt[c];
      H16[w][c] = (_Float16)fmaxf(h, 0.f);
    }
  }
  __syncthreads();

  // Stage B: out = H @ w2.T  (M=16 padded, N=1024, K=1024)
  for (int t = w; t < DV / 16; t += 8) {
    v8f acc = {};
    for (int kc = 0; kc < DV / 32; ++kc) {
      v16h af = a_frag_lds(&H16[0][0], DV, kc * 32);
      v16h bf = b_frag_w(w2, DV, t * 16 + (lane & 15), kc * 32);
      acc = __builtin_amdgcn_wmma_f32_16x16x32_f16(false, af, false, bf, (short)0, acc, false, false);
    }
    if (lane < 16) {
      int n = t * 16 + lane;
      float bias = b2[n];
#pragma unroll
      for (int j = 0; j < 8; ++j) {
        float v = acc[j] + bias;
        if (sigma_mode) {
          float sp = fmaxf(v, 0.f) + log1pf(expf(-fabsf(v))) + 1e-6f;
          out[j * DV + n] = sp;
          wsv[j * DV + n] = 1.f / sp;
        } else {
          out[j * DV + n] = v;
          wsv[j * DV + n] = v;
        }
      }
    }
  }
}

// ----------------------------------------- kernel 2: streaming Mahalanobis
// One wave per (b,t) row; 8 rows per block; float4 coalesced streaming.
__global__ void __launch_bounds__(256)
dist_kernel(const float* __restrict__ video,
            const float* __restrict__ mu,    // ws [8, DV]
            const float* __restrict__ isg,   // ws [8, DV] = 1/sigma
            float* __restrict__ dist) {      // [8, T]
  int w = threadIdx.x >> 5, lane = threadIdx.x & 31;
  int row = blockIdx.x * 8 + w;   // 0 .. B*T-1
  int b = row >> 13;
  const float4* v4 = (const float4*)(video + (size_t)row * DV);
  const float4* m4 = (const float4*)(mu + b * DV);
  const float4* s4 = (const float4*)(isg + b * DV);
  float s = 0.f;
  for (int i = lane; i < DV / 4; i += 32) {
    __builtin_prefetch((const void*)(v4 + i + 256), 0, 1);  // next row, speculative
    float4 f = v4[i]; float4 m = m4[i]; float4 iv = s4[i];
    float dx = f.x - m.x, dy = f.y - m.y, dz = f.z - m.z, dw = f.w - m.w;
    s += dx * dx * iv.x + dy * dy * iv.y + dz * dz * iv.z + dw * dw * iv.w;
  }
  s = wave_sum(s);
  if (lane == 0) dist[row] = s;
}

// ----------------------------- kernel 3: median / top-18 / diversify (sort)
__device__ __forceinline__ void bitonic_sort_8192(unsigned long long* key, int tid) {
  for (unsigned k = 2; k <= 8192u; k <<= 1) {
    for (unsigned j = k >> 1; j > 0; j >>= 1) {
#pragma unroll
      for (int e = 0; e < 8; ++e) {
        unsigned i = (unsigned)tid + (unsigned)e * 1024u;
        unsigned ixj = i ^ j;
        if (ixj > i) {
          unsigned long long a = key[i], q = key[ixj];
          bool up = ((i & k) == 0);
          if ((a > q) == up) { key[i] = q; key[ixj] = a; }
        }
      }
      __syncthreads();
    }
  }
}

__global__ void __launch_bounds__(1024)
select_kernel(const float* __restrict__ dist,  // [8, T]
              float* __restrict__ idx_out,     // [8, KREP] as float
              int* __restrict__ idx_ws) {      // [8, KREP] ints
  __shared__ unsigned long long key[T_];       // 64 KB
  int b = blockIdx.x;
  int tid = threadIdx.x;
  const float* d = dist + b * T_;

  // (dist, idx) keys; dist >= 0 so float bits are monotonic as uint
  for (int e = 0; e < 8; ++e) {
    int i = tid + e * 1024;
    key[i] = (((unsigned long long)__float_as_uint(d[i])) << 32) | (unsigned)i;
  }
  __syncthreads();
  bitonic_sort_8192(key, tid);
  float med = __uint_as_float((unsigned)(key[(T_ - 1) / 2] >> 32));  // lower median
  __syncthreads();

  // (|d - med|, idx) keys -> first CAND after sort == top_k order with ties
  for (int e = 0; e < 8; ++e) {
    int i = tid + e * 1024;
    key[i] = (((unsigned long long)__float_as_uint(fabsf(d[i] - med))) << 32) | (unsigned)i;
  }
  __syncthreads();
  bitonic_sort_8192(key, tid);

  if (tid == 0) {
    int cand[CAND]; float mind[CAND];
#pragma unroll
    for (int j = 0; j < CAND; ++j) cand[j] = (int)(unsigned)(key[j] & 0xffffffffu);
    float c0 = (float)cand[0];
#pragma unroll
    for (int j = 0; j < CAND; ++j) mind[j] = fabsf((float)cand[j] - c0);
    mind[0] = -__builtin_inff();
    int sel[KREP]; sel[0] = cand[0];
    for (int it = 1; it < KREP; ++it) {
      int best = 0; float bv = mind[0];
      for (int j = 1; j < CAND; ++j) if (mind[j] > bv) { bv = mind[j]; best = j; }
      sel[it] = cand[best];
      float cb = (float)cand[best];
      for (int j = 0; j < CAND; ++j) mind[j] = fminf(mind[j], fabsf((float)cand[j] - cb));
      mind[best] = -__builtin_inff();
    }
#pragma unroll
    for (int j = 0; j < KREP; ++j) {
      idx_out[b * KREP + j] = (float)sel[j];
      idx_ws[b * KREP + j]  = sel[j];
    }
  }
}

// ---------------------------------------- kernel 4: gather + MHA (WMMA)
// abuf: zero-padded 16-row A buffer, holds rep for GEMM1, attn-out for GEMM2.
__global__ void __launch_bounds__(256)
mha_kernel(const float* __restrict__ video,
           const int* __restrict__ idx_ws,   // [8, KREP]
           const float* __restrict__ in_w,   // [3*DV, DV]
           const float* __restrict__ in_b,   // [3*DV]
           const float* __restrict__ out_w,  // [DV, DV]
           const float* __restrict__ out_b,  // [DV]
           float* __restrict__ refined,      // [8, KREP, DV]
           float* __restrict__ attn_out) {   // [8, KREP, KREP]
  __shared__ __align__(16) _Float16 abuf[16][DV];      // 32 KB
  __shared__ __align__(16) _Float16 qkv16[KREP][3 * DV]; // 36 KB
  __shared__ float attn_s[NHEADS][KREP][KREP];
  int b = blockIdx.x;
  int tid = threadIdx.x;
  int w = tid >> 5, lane = tid & 31;

  // zero pad rows 6..15, gather selected rows 0..5 as f16
  for (int i = tid; i < 10 * DV; i += 256) (&abuf[KREP][0])[i] = (_Float16)0.f;
  for (int m = 0; m < KREP; ++m) {
    int t = idx_ws[b * KREP + m];
    const float* src = video + ((size_t)(b * T_ + t)) * DV;
    for (int c = tid; c < DV; c += 256) abuf[m][c] = (_Float16)src[c];
  }
  __syncthreads();

  // qkv = rep @ in_w.T + in_b  (M=16 padded, N=3072 -> 192 tiles, K=1024)
  for (int t = w; t < 192; t += 8) {
    v8f acc = {};
    for (int kc = 0; kc < DV / 32; ++kc) {
      v16h af = a_frag_lds(&abuf[0][0], DV, kc * 32);
      v16h bf = b_frag_w(in_w, DV, t * 16 + (lane & 15), kc * 32);
      acc = __builtin_amdgcn_wmma_f32_16x16x32_f16(false, af, false, bf, (short)0, acc, false, false);
    }
    if (lane < 16) {
      int n = t * 16 + lane;
      float bias = in_b[n];
#pragma unroll
      for (int j = 0; j < 8; ++j)
        if (j < KREP) qkv16[j][n] = (_Float16)(acc[j] + bias);
    }
  }
  __syncthreads();

  // attention: wave w == head w; writes attn-out into abuf rows 0..5
  {
    int h = w;
    const _Float16* q  = &qkv16[0][h * HD];
    const _Float16* kk = &qkv16[0][DV + h * HD];
    const _Float16* vv = &qkv16[0][2 * DV + h * HD];
    float sc[KREP][KREP];
#pragma unroll
    for (int i = 0; i < KREP; ++i)
#pragma unroll
      for (int j = 0; j < KREP; ++j) {
        float s = 0.f;
        for (int dd = lane; dd < HD; dd += 32)
          s += (float)q[i * 3 * DV + dd] * (float)kk[j * 3 * DV + dd];
        s = wave_sum(s);
        sc[i][j] = s * 0.0883883476483184f;  // 1/sqrt(128)
      }
#pragma unroll
    for (int i = 0; i < KREP; ++i) {
      float mx = sc[i][0];
#pragma unroll
      for (int j = 1; j < KREP; ++j) mx = fmaxf(mx, sc[i][j]);
      float sum = 0.f;
#pragma unroll
      for (int j = 0; j < KREP; ++j) { sc[i][j] = expf(sc[i][j] - mx); sum += sc[i][j]; }
      float inv = 1.f / sum;
#pragma unroll
      for (int j = 0; j < KREP; ++j) sc[i][j] *= inv;
      if (lane == 0)
        for (int j = 0; j < KREP; ++j) attn_s[h][i][j] = sc[i][j];
    }
#pragma unroll
    for (int i = 0; i < KREP; ++i)
      for (int dd = lane; dd < HD; dd += 32) {
        float o = 0.f;
#pragma unroll
        for (int j = 0; j < KREP; ++j) o += sc[i][j] * (float)vv[j * 3 * DV + dd];
        abuf[i][h * HD + dd] = (_Float16)o;  // rep is dead; rows 6..15 stay zero
      }
  }
  __syncthreads();

  // attn weights averaged over heads
  if (tid < KREP * KREP) {
    float s = 0.f;
    int i = tid / KREP, j = tid % KREP;
    for (int h = 0; h < NHEADS; ++h) s += attn_s[h][i][j];
    attn_out[b * KREP * KREP + tid] = s * (1.f / NHEADS);
  }

  // refined = ao @ out_w.T + out_b  (N=1024 -> 64 tiles, K=1024)
  for (int t = w; t < 64; t += 8) {
    v8f acc = {};
    for (int kc = 0; kc < DV / 32; ++kc) {
      v16h af = a_frag_lds(&abuf[0][0], DV, kc * 32);
      v16h bf = b_frag_w(out_w, DV, t * 16 + (lane & 15), kc * 32);
      acc = __builtin_amdgcn_wmma_f32_16x16x32_f16(false, af, false, bf, (short)0, acc, false, false);
    }
    if (lane < 16) {
      int n = t * 16 + lane;
      float bias = out_b[n];
#pragma unroll
      for (int j = 0; j < 8; ++j)
        if (j < KREP) refined[((size_t)b * KREP + j) * DV + n] = acc[j] + bias;
    }
  }
}

// ----------------------------------------------------------------- launcher
extern "C" void kernel_launch(void* const* d_in, const int* in_sizes, int n_in,
                              void* d_out, int out_size, void* d_ws, size_t ws_size,
                              hipStream_t stream) {
  const float* video = (const float*)d_in[0];
  const float* query = (const float*)d_in[1];
  const float* mu_w1 = (const float*)d_in[2];
  const float* mu_b1 = (const float*)d_in[3];
  const float* mu_g  = (const float*)d_in[4];
  const float* mu_bt = (const float*)d_in[5];
  const float* mu_w2 = (const float*)d_in[6];
  const float* mu_b2 = (const float*)d_in[7];
  const float* sg_w1 = (const float*)d_in[8];
  const float* sg_b1 = (const float*)d_in[9];
  const float* sg_g  = (const float*)d_in[10];
  const float* sg_bt = (const float*)d_in[11];
  const float* sg_w2 = (const float*)d_in[12];
  const float* sg_b2 = (const float*)d_in[13];
  const float* in_w  = (const float*)d_in[14];
  const float* in_b  = (const float*)d_in[15];
  const float* out_w = (const float*)d_in[16];
  const float* out_b = (const float*)d_in[17];

  float* out = (float*)d_out;
  float* ws_f   = (float*)d_ws;
  float* ws_mu  = ws_f;                        // [8, DV]
  float* ws_isg = ws_f + B_ * DV;              // [8, DV]  (1/sigma)
  int*   ws_idx = (int*)(ws_f + 2 * B_ * DV);  // [8, KREP]

  // MLP branches (WMMA)
  mlp_ln_wmma<<<1, 256, 0, stream>>>(query, mu_w1, mu_b1, mu_g, mu_bt, mu_w2, mu_b2,
                                     out + MU_OFF, ws_mu, 0);
  mlp_ln_wmma<<<1, 256, 0, stream>>>(query, sg_w1, sg_b1, sg_g, sg_bt, sg_w2, sg_b2,
                                     out + SG_OFF, ws_isg, 1);
  // HBM-bound Mahalanobis streaming pass
  dist_kernel<<<(B_ * T_) / 8, 256, 0, stream>>>(video, ws_mu, ws_isg, out + DIST_OFF);
  // per-batch median/top-k/diversify via LDS bitonic sort
  select_kernel<<<B_, 1024, 0, stream>>>(out + DIST_OFF, out + IDX_OFF, ws_idx);
  // gather + MHA (WMMA)
  mha_kernel<<<B_, 256, 0, stream>>>(video, ws_idx, in_w, in_b, out_w, out_b,
                                     out + REFINED_OFF, out + ATTN_OFF);
}
